// TprRnn_42245298323612
// MI455X (gfx1250) — matrix operand order repaired and checked by reading, not compiled, not run on hardware
//
#include <hip/hip_runtime.h>

typedef __attribute__((ext_vector_type(16))) __bf16 v16bf;
typedef __attribute__((ext_vector_type(8)))  __bf16 v8bf;
typedef __attribute__((ext_vector_type(8)))  float  v8f;
typedef __attribute__((ext_vector_type(4)))  float  v4f;
typedef __attribute__((ext_vector_type(4)))  unsigned int ui32x4;
typedef __attribute__((ext_vector_type(8)))  int i32x8;
typedef __attribute__((ext_vector_type(4)))  int i32x4;

#define BS   4096      // B*T tokens
#define D_   512       // input dim
#define KD   256       // k_dim
#define VD   512       // v_dim
#define S_   4         // slots
#define NK   16384     // keys per slot
#define TOPK 32
#define MT   64        // query rows per score workgroup
#define CK   32        // keys per streamed chunk

#if defined(__has_builtin)
#  if __has_builtin(__builtin_amdgcn_tensor_load_to_lds)
#    define USE_TDM 1
#  endif
#endif
#ifndef USE_TDM
#  define USE_TDM 0
#endif

// ---- WMMA A-fragment loader (16x32 bf16, row-major source) -----------------
// lane (m = l&15, hf = l>>4); elem e -> k = 16*hf + e (contiguous, matches B),
// loaded as two contiguous 16-byte runs per lane.
__device__ inline v16bf load_a_frag(const __bf16* __restrict__ row, int kbase, int hf) {
    union { v16bf v; v8bf h[2]; } u;
    u.h[0] = *(const v8bf*)(row + kbase + 16 * hf);
    u.h[1] = *(const v8bf*)(row + kbase + 16 * hf + 8);
    return u.v;
}

#if USE_TDM
// ---- TDM descriptor: load CK rows x KD bf16 (row stride KD) into LDS -------
// Packing per CDNA5 ISA (2-D tile, trailing
// groups zero / unused). data_size=1 (2 bytes). Tensor dims only bound OOB
// (never hit). This toolchain exposes the 6-arg builtin:
//   (uint32x4 g0, int32x8 g1, int32x4, int32x4, int32x8, i32 cpol)
__device__ inline void tdm_load_chunk(const __bf16* gsrc, unsigned lds_off) {
    unsigned long long ga = (unsigned long long)(const void*)gsrc;
    ui32x4 g0;
    g0[0] = 1u;                                   // count=1, user mode
    g0[1] = lds_off;                              // lds_addr
    g0[2] = (unsigned)ga;                         // global_addr[31:0]
    g0[3] = (unsigned)((ga >> 32) & 0x1FFFFFFu) | (2u << 30);  // addr[56:32] | type=2
    i32x8 g1;
    g1[0] = 0x10000;                              // data_size=1 (2B), no pad/iter
    g1[1] = (int)((unsigned)(KD & 0xFFFF) << 16); // tensor_dim0[15:0] @bit48
    g1[2] = (int)(((unsigned)KD >> 16) | ((unsigned)(NK & 0xFFFF) << 16)); // dim0 hi | dim1 lo
    g1[3] = (int)(((unsigned)NK >> 16) | ((unsigned)KD << 16));            // dim1 hi | tile_dim0
    g1[4] = CK;                                   // tile_dim1 | tile_dim2=0
    g1[5] = KD;                                   // tensor_dim0_stride[31:0]
    g1[6] = 0;
    g1[7] = 0;
    i32x4 gz4 = {0, 0, 0, 0};
    i32x8 gz8 = {0, 0, 0, 0, 0, 0, 0, 0};
    __builtin_amdgcn_tensor_load_to_lds(g0, g1, gz4, gz4, gz8, 0);
}
#endif

// ---- fp32 -> bf16 elementwise ---------------------------------------------
__global__ __launch_bounds__(256)
void cvt_bf16_kernel(const float* __restrict__ in, __bf16* __restrict__ out, int n) {
    int i = blockIdx.x * 256 + threadIdx.x;
    if (i < n) out[i] = (__bf16)in[i];
}

// ---- fp32 (rows x cols) -> bf16 transposed (cols x rows) -------------------
__global__ __launch_bounds__(256)
void transpose_bf16_kernel(const float* __restrict__ in, __bf16* __restrict__ out,
                           int rows, int cols) {
    int i = blockIdx.x * 256 + threadIdx.x;
    if (i < rows * cols) {
        int r = i / cols, c = i - r * cols;
        out[(size_t)c * rows + r] = (__bf16)in[i];
    }
}

// ---- normalize key rows, emit bf16 (one wave32 per 256-float row) ----------
__global__ __launch_bounds__(256)
void normkeys_kernel(const float* __restrict__ keys, __bf16* __restrict__ knbf) {
    int row  = blockIdx.x * 8 + (threadIdx.x >> 5);   // 65536 rows total
    int lane = threadIdx.x & 31;
    const float* kr = keys + (size_t)row * KD + lane * 8;
    float v[8]; float ss = 0.f;
#pragma unroll
    for (int j = 0; j < 8; ++j) { v[j] = kr[j]; ss += v[j] * v[j]; }
#pragma unroll
    for (int off = 16; off > 0; off >>= 1) ss += __shfl_xor(ss, off, 32);
    float rs = rsqrtf(ss);
    __bf16* orow = knbf + (size_t)row * KD + lane * 8;
#pragma unroll
    for (int j = 0; j < 8; ++j) orow[j] = (__bf16)(v[j] * rs);
}

// ---- q = x @ Wq + bq  (M=4096, N=256, K=512), WMMA bf16, bias in acc -------
__global__ __launch_bounds__(128)
void qproj_kernel(const __bf16* __restrict__ xbf, const __bf16* __restrict__ WqT,
                  const float* __restrict__ bq, __bf16* __restrict__ qb16) {
    const int mt = blockIdx.x;                 // 0..255
    const int tid = threadIdx.x, lane = tid & 31, w = tid >> 5;
    const int m = lane & 15, hf = lane >> 4;
    const int n0 = (blockIdx.y * 4 + w) * 16;  // gridDim.y = 4 -> N = 256

    const __bf16* arow = xbf + (size_t)(mt * 16 + m) * D_;
    const __bf16* brow = WqT + (size_t)(n0 + m) * D_ + 16 * hf;

    float bias = bq[n0 + m];
    v8f acc = {bias, bias, bias, bias, bias, bias, bias, bias};
#pragma unroll
    for (int kb = 0; kb < 16; ++kb) {
        v16bf a = load_a_frag(arow, kb * 32, hf);
        v16bf b = *(const v16bf*)(brow + kb * 32);
        acc = __builtin_amdgcn_wmma_f32_16x16x32_bf16(false, a, false, b,
                                                      (short)0, acc, false, false);
    }
#pragma unroll
    for (int r = 0; r < 8; ++r) {
        int row = mt * 16 + r + 8 * hf;
        qb16[(size_t)row * KD + (n0 + m)] = (__bf16)acc[r];
    }
}

// ---- res = q @ Wr + br  (M=4096, N=512, K=256), WMMA bf16 ------------------
__global__ __launch_bounds__(128)
void resid_kernel(const __bf16* __restrict__ qb16, const __bf16* __restrict__ WrT,
                  const float* __restrict__ br, float* __restrict__ res) {
    const int mt = blockIdx.x;
    const int tid = threadIdx.x, lane = tid & 31, w = tid >> 5;
    const int m = lane & 15, hf = lane >> 4;
    const int n0 = (blockIdx.y * 4 + w) * 16;  // gridDim.y = 8 -> N = 512

    const __bf16* arow = qb16 + (size_t)(mt * 16 + m) * KD;
    const __bf16* brow = WrT + (size_t)(n0 + m) * KD + 16 * hf;

    float bias = br[n0 + m];
    v8f acc = {bias, bias, bias, bias, bias, bias, bias, bias};
#pragma unroll
    for (int kb = 0; kb < 8; ++kb) {
        v16bf a = load_a_frag(arow, kb * 32, hf);
        v16bf b = *(const v16bf*)(brow + kb * 32);
        acc = __builtin_amdgcn_wmma_f32_16x16x32_bf16(false, a, false, b,
                                                      (short)0, acc, false, false);
    }
#pragma unroll
    for (int r = 0; r < 8; ++r) {
        int row = mt * 16 + r + 8 * hf;
        res[(size_t)row * VD + (n0 + m)] = acc[r];
    }
}

// ---- fused scores GEMM + streaming top-k ----------------------------------
// Grid: (BS/MT, S_), block = 256 (8 waves). Each WG: 64 query rows x 1 slot.
// Wave w: M-tile mq = w&3 (rows mq*16..), N-half nh = w>>2 of the 32-key chunk.
// Keys stream global -> LDS via TDM double-buffer (TENSORcnt); B fragments are
// ds_load'd with 4-way reuse. A fragments live in VGPRs for the whole stream.
__global__ __launch_bounds__(256)
void score_topk_kernel(const __bf16* __restrict__ qbf, const __bf16* __restrict__ knbf,
                       float* __restrict__ topS, int* __restrict__ topI) {
    const int qb = blockIdx.x;          // 0..63 (64-row tiles)
    const int s  = blockIdx.y;          // 0..3
    const int tid = threadIdx.x, lane = tid & 31, w = tid >> 5;
    const int mq = w & 3, nh = w >> 2;
    const int m = lane & 15, hf = lane >> 4;

    __shared__ __bf16 keybuf[2][CK * KD];   // 2 x 16KB double buffer
    __shared__ float sc[MT][CK + 4];        // chunk scores (padded)
    __shared__ float topv[MT][TOPK];
    __shared__ int   topi[MT][TOPK];
    __shared__ float rowminv[MT];
    __shared__ int   rowminp[MT];
    __shared__ float segmax[MT][4];

    for (int i = tid; i < MT * TOPK; i += 256) {
        topv[i >> 5][i & 31] = -3.0e38f;
        topi[i >> 5][i & 31] = 0;
    }
    if (tid < MT) { rowminv[tid] = -3.0e38f; rowminp[tid] = 0; }

    // A fragments: this wave's 16 query rows x 256 K, resident in VGPRs.
    const __bf16* qrow = qbf + (size_t)(qb * MT + mq * 16 + m) * KD;
    v16bf A[8];
#pragma unroll
    for (int kb = 0; kb < 8; ++kb) A[kb] = load_a_frag(qrow, kb * 32, hf);

    const __bf16* knS = knbf + (size_t)s * NK * KD;

#if USE_TDM
    if (w == 0)   // TDM: EXEC-independent, one issue per WG; tracked by TENSORcnt
        tdm_load_chunk(knS, (unsigned)(unsigned long long)(void*)&keybuf[0][0]);
#endif
    __syncthreads();

    for (int c = 0; c < NK / CK; ++c) {
        const int cur = c & 1;
#if USE_TDM
        if (w == 0) __builtin_amdgcn_s_wait_tensorcnt(0);   // chunk c landed
        __syncthreads();
        if (w == 0 && c + 1 < NK / CK)                      // overlap copy c+1
            tdm_load_chunk(knS + (size_t)(c + 1) * CK * KD,
                           (unsigned)(unsigned long long)(void*)&keybuf[cur ^ 1][0]);
#else
        // cooperative single-buffer copy: 256 threads x 4 x 16B = 16KB
        for (int i = tid; i < CK * KD / 8; i += 256)
            ((v8bf*)keybuf[0])[i] = ((const v8bf*)(knS + (size_t)c * CK * KD))[i];
        __syncthreads();
#endif
        // wave computes its 16x16 tile: keys [c*CK + nh*16, +16)
        {
            const __bf16* brow = &keybuf[USE_TDM ? cur : 0][(nh * 16 + m) * KD + 16 * hf];
            v8f acc = {};
#pragma unroll
            for (int kb = 0; kb < 8; ++kb) {
                v16bf b = *(const v16bf*)(brow + kb * 32);   // ds_load_b128 x2
                acc = __builtin_amdgcn_wmma_f32_16x16x32_bf16(false, A[kb], false, b,
                                                              (short)0, acc, false, false);
            }
#pragma unroll
            for (int r = 0; r < 8; ++r)
                sc[mq * 16 + r + 8 * hf][nh * 16 + m] = acc[r];
        }
        __syncthreads();

        // candidate filter: 4 threads per row over 8 scores each
        {
            const int r = tid >> 2, seg = tid & 3;
            float mx = -3.0e38f;
#pragma unroll
            for (int j = 0; j < 8; ++j) mx = fmaxf(mx, sc[r][seg * 8 + j]);
            segmax[r][seg] = mx;
        }
        __syncthreads();

        // rare path: per-row serial insert (expected ~200 inserts over 16384 keys)
        if (tid < MT) {
            const int r = tid;
            float mx = fmaxf(fmaxf(segmax[r][0], segmax[r][1]),
                             fmaxf(segmax[r][2], segmax[r][3]));
            if (mx > rowminv[r]) {
                float mn = rowminv[r]; int mp = rowminp[r];
                for (int j = 0; j < CK; ++j) {
                    float v = sc[r][j];
                    if (v > mn) {
                        topv[r][mp] = v; topi[r][mp] = c * CK + j;
                        mn = topv[r][0]; mp = 0;
#pragma unroll
                        for (int q2 = 1; q2 < TOPK; ++q2)
                            if (topv[r][q2] < mn) { mn = topv[r][q2]; mp = q2; }
                    }
                }
                rowminv[r] = mn; rowminp[r] = mp;
            }
        }
        __syncthreads();
    }

    for (int i = tid; i < MT * TOPK; i += 256) {
        int r = i >> 5, k = i & 31;
        size_t o = ((size_t)s * BS + qb * MT + r) * TOPK + k;
        topS[o] = topv[r][k];
        topI[o] = topi[r][k];
    }
}

// ---- softmax over top-32 + weighted value gather + residual add ------------
__global__ __launch_bounds__(128)
void gather_kernel(const float* __restrict__ values, const float* __restrict__ topS,
                   const int* __restrict__ topI, const float* __restrict__ res,
                   float* __restrict__ out) {
    const int b = blockIdx.x, s = blockIdx.y, tid = threadIdx.x;
    __shared__ float wts[TOPK];
    __shared__ int   idxs[TOPK];
    if (tid < TOPK) {
        size_t o = ((size_t)s * BS + b) * TOPK + tid;
        float v = topS[o];
        idxs[tid] = topI[o];
        float mx = v;
#pragma unroll
        for (int off = 16; off > 0; off >>= 1) mx = fmaxf(mx, __shfl_xor(mx, off, 32));
        float e = __expf(v - mx);
        float sm = e;
#pragma unroll
        for (int off = 16; off > 0; off >>= 1) sm += __shfl_xor(sm, off, 32);
        wts[tid] = e / sm;
    }
    __syncthreads();
    const int col = tid * 4;                       // 128 threads x float4 = 512
    v4f acc = *(const v4f*)(res + (size_t)b * VD + col);
    const float* vs = values + (size_t)s * NK * VD;
    for (int k = 0; k < TOPK; ++k) {
        v4f vv = *(const v4f*)(vs + (size_t)idxs[k] * VD + col);
        acc += wts[k] * vv;
    }
    *(v4f*)(out + ((size_t)b * S_ + s) * VD + col) = acc;
}

// ---- workspace layout (bytes, 256-aligned) ---------------------------------
#define WS_QBF   0u          // 4096*256 bf16          = 2,097,152
#define WS_XBF   2097152u    // 4096*512 bf16          = 4,194,304
#define WS_WQT   6291456u    // 256*512 bf16 (n-major) =   262,144
#define WS_WRT   6553600u    // 512*256 bf16 (n-major) =   262,144
#define WS_KNBF  6815744u    // 4*16384*256 bf16       = 33,554,432
#define WS_TOPS  40370176u   // 4*4096*32 f32          = 2,097,152
#define WS_TOPI  42467328u   // 4*4096*32 i32          = 2,097,152
#define WS_RES   44564480u   // 4096*512 f32           = 8,388,608
                             // total                  = 52,953,088

extern "C" void kernel_launch(void* const* d_in, const int* in_sizes, int n_in,
                              void* d_out, int out_size, void* d_ws, size_t ws_size,
                              hipStream_t stream) {
    const float* x      = (const float*)d_in[0];
    const float* Wq     = (const float*)d_in[1];
    const float* bq     = (const float*)d_in[2];
    const float* keys   = (const float*)d_in[3];
    const float* values = (const float*)d_in[4];
    const float* Wr     = (const float*)d_in[5];
    const float* br     = (const float*)d_in[6];
    float* out = (float*)d_out;

    char* ws = (char*)d_ws;
    __bf16* qb16 = (__bf16*)(ws + WS_QBF);
    __bf16* xbf  = (__bf16*)(ws + WS_XBF);
    __bf16* WqT  = (__bf16*)(ws + WS_WQT);
    __bf16* WrT  = (__bf16*)(ws + WS_WRT);
    __bf16* knbf = (__bf16*)(ws + WS_KNBF);
    float*  topS = (float*)(ws + WS_TOPS);
    int*    topI = (int*)(ws + WS_TOPI);
    float*  res  = (float*)(ws + WS_RES);

    cvt_bf16_kernel<<<(BS * D_ + 255) / 256, 256, 0, stream>>>(x, xbf, BS * D_);
    transpose_bf16_kernel<<<(D_ * KD + 255) / 256, 256, 0, stream>>>(Wq, WqT, D_, KD);
    transpose_bf16_kernel<<<(KD * VD + 255) / 256, 256, 0, stream>>>(Wr, WrT, KD, VD);
    normkeys_kernel<<<(S_ * NK) / 8, 256, 0, stream>>>(keys, knbf);

    qproj_kernel<<<dim3(BS / 16, KD / 64), 128, 0, stream>>>(xbf, WqT, bq, qb16);
    resid_kernel<<<dim3(BS / 16, VD / 64), 128, 0, stream>>>(qb16, WrT, br, res);
    score_topk_kernel<<<dim3(BS / MT, S_), 256, 0, stream>>>(qb16, knbf, topS, topI);
    gather_kernel<<<dim3(BS, S_), 128, 0, stream>>>(values, topS, topI, res, out);
}
